// CausalSelfAttention_50302656971480
// MI455X (gfx1250) — matrix operand order repaired and checked by reading, not compiled, not run on hardware
//
#include <hip/hip_runtime.h>

// ---------------------------------------------------------------------------
// CDNA5 (gfx1250, wave32) causal self-attention.
// B=4, S=2048, D=1024, H=16, Dh=64. All math on v_wmma_f32_16x16x32_bf16.
// Round 3: fix async-builtin pointer types (AS1/AS3 generic-vector int4).
// ---------------------------------------------------------------------------

#define ATT_B   4
#define ATT_S   2048
#define ATT_D   1024
#define ATT_H   16
#define ATT_DH  64
#define ROWS    (ATT_B * ATT_S)          // 8192

typedef __attribute__((ext_vector_type(16))) __bf16 bf16x16;
typedef __attribute__((ext_vector_type(8)))  float  floatx8;
typedef int vec4i __attribute__((vector_size(16)));   // matches builtin proto

__device__ __forceinline__ unsigned short f32_to_bf16(float f) {
  unsigned int u = __float_as_uint(f);
  u += 0x7FFFu + ((u >> 16) & 1u);     // round-to-nearest-even
  return (unsigned short)(u >> 16);
}

union FragCast { bf16x16 v; uint4 q[2]; };

__device__ __forceinline__ bf16x16 make_frag(uint4 lo, uint4 hi) {
  FragCast f; f.q[0] = lo; f.q[1] = hi; return f.v;
}

__device__ __forceinline__ floatx8 wmma_bf16(bf16x16 a, bf16x16 b, floatx8 c) {
  return __builtin_amdgcn_wmma_f32_16x16x32_bf16(false, a, false, b, (short)0, c,
                                                 false, false);
}

// ---- async global->LDS (gfx1250) with safe fallback -----------------------
#if __has_builtin(__builtin_amdgcn_global_load_async_to_lds_b128)
#define HAVE_ASYNC_LDS 1
#else
#define HAVE_ASYNC_LDS 0
#endif

__device__ __forceinline__ void async_copy_b128(const unsigned short* src,
                                                unsigned short* dst_lds) {
#if HAVE_ASYNC_LDS
  __builtin_amdgcn_global_load_async_to_lds_b128(
      (__attribute__((address_space(1))) vec4i*)src,
      (__attribute__((address_space(3))) vec4i*)dst_lds, 0, 0);
#else
  *(uint4*)dst_lds = *(const uint4*)src;
#endif
}

__device__ __forceinline__ void wait_async_zero() {
#if HAVE_ASYNC_LDS
#if __has_builtin(__builtin_amdgcn_s_wait_asynccnt)
  __builtin_amdgcn_s_wait_asynccnt(0);
#else
  asm volatile("s_wait_asynccnt 0x0" ::: "memory");
#endif
#endif
}

// ======================= Kernel 1: fused QKV projection ====================
// C = x @ W + bias for W in {Wq,Wk,Wv} selected by blockIdx.y.
// Block tile 128x128, 8 waves of 32x64 each, K-step 32 (one WMMA depth).
#define LDSTR 40   // ushorts per LDS row (32 data + 8 pad)

__global__ __launch_bounds__(256) void qkv_gemm_kernel(
    const float* __restrict__ x,
    const float* __restrict__ Wq, const float* __restrict__ bq,
    const float* __restrict__ Wk, const float* __restrict__ bk,
    const float* __restrict__ Wv, const float* __restrict__ bv,
    unsigned short* __restrict__ q_ws,   // (B,H,S,Dh) bf16, pre-scaled 1/sqrt(Dh)
    unsigned short* __restrict__ k_ws,   // (B,H,S,Dh) bf16
    unsigned short* __restrict__ vt_ws)  // (B,H,Dh,S) bf16 (transposed V)
{
  __shared__ unsigned short lAs[128 * LDSTR];  // A tile [m][k] bf16
  __shared__ unsigned short lBs[128 * LDSTR];  // B tile transposed [n][k] bf16

  const int tid   = threadIdx.x;
  const int lane  = tid & 31;
  const int wave  = tid >> 5;
  const int waveM = wave & 3;           // 4 wave rows
  const int waveN = wave >> 2;          // 2 wave cols
  const int half  = lane >> 4;          // lane half select
  const int l16   = lane & 15;

  const int mBase = blockIdx.x * 128;
  const int gN    = blockIdx.y * 128;   // 0..3071 across [q|k|v]
  const int mat   = gN >> 10;           // 0=q 1=k 2=v
  const int nBase = gN & (ATT_D - 1);

  const float* W    = (mat == 0) ? Wq : (mat == 1) ? Wk : Wv;
  const float* bias = (mat == 0) ? bq : (mat == 1) ? bk : bv;

  floatx8 acc[2][4];
#pragma unroll
  for (int i = 0; i < 2; ++i)
#pragma unroll
    for (int j = 0; j < 4; ++j) acc[i][j] = (floatx8){0,0,0,0,0,0,0,0};

  // per-thread staging coordinates (reused every K-step)
  const int rowA0 = tid >> 3, c4A = (tid & 7) << 2;     // A chunk for i==0
  const int rowB0 = tid >> 5, c4B = (tid & 31) << 2;    // B chunk for i==0

  for (int k0 = 0; k0 < ATT_D; k0 += 32) {
    // ---- stage A: 128x32 f32 -> bf16 ----
#pragma unroll
    for (int i = 0; i < 4; ++i) {
      int idx = tid + 256 * i;               // 0..1023 float4 chunks
      int row = idx >> 3;
      int c4  = (idx & 7) << 2;
      float4 f = *(const float4*)(x + (size_t)(mBase + row) * ATT_D + k0 + c4);
      uint2 p;
      p.x = (unsigned)f32_to_bf16(f.x) | ((unsigned)f32_to_bf16(f.y) << 16);
      p.y = (unsigned)f32_to_bf16(f.z) | ((unsigned)f32_to_bf16(f.w) << 16);
      *(uint2*)(&lAs[row * LDSTR + c4]) = p;
    }
    // ---- stage B transposed: W[k0..+31][nBase..+127] -> [n][k] ----
#pragma unroll
    for (int i = 0; i < 4; ++i) {
      int idx = tid + 256 * i;
      int row = idx >> 5;                    // k row 0..31
      int c4  = (idx & 31) << 2;             // n col
      float4 f = *(const float4*)(W + (size_t)(k0 + row) * ATT_D + nBase + c4);
      lBs[(c4 + 0) * LDSTR + row] = f32_to_bf16(f.x);
      lBs[(c4 + 1) * LDSTR + row] = f32_to_bf16(f.y);
      lBs[(c4 + 2) * LDSTR + row] = f32_to_bf16(f.z);
      lBs[(c4 + 3) * LDSTR + row] = f32_to_bf16(f.w);
    }
    // hint next K-step tiles into cache while we compute this one
    if (k0 + 32 < ATT_D) {
      __builtin_prefetch(x + (size_t)(mBase + rowA0) * ATT_D + k0 + 32 + c4A, 0, 1);
      __builtin_prefetch(W + (size_t)(k0 + 32 + rowB0) * ATT_D + nBase + c4B, 0, 1);
    }
    __syncthreads();

    bf16x16 af[2], bf[4];
    const int kbA = half * 16;
#pragma unroll
    for (int mt = 0; mt < 2; ++mt) {
      const unsigned short* pa = &lAs[(waveM * 32 + mt * 16 + l16) * LDSTR];
      af[mt] = make_frag(*(const uint4*)(pa + kbA),
                         *(const uint4*)(pa + kbA + 8));
    }
    const int kbB = half * 16;
#pragma unroll
    for (int nt = 0; nt < 4; ++nt) {
      const unsigned short* pb = &lBs[(waveN * 64 + nt * 16 + l16) * LDSTR + kbB];
      bf[nt] = make_frag(*(const uint4*)pb, *(const uint4*)(pb + 8));
    }
#pragma unroll
    for (int mt = 0; mt < 2; ++mt)
#pragma unroll
      for (int nt = 0; nt < 4; ++nt)
        acc[mt][nt] = wmma_bf16(af[mt], bf[nt], acc[mt][nt]);
    __syncthreads();
  }

  // ---- epilogue: scatter into attention layouts ----
#pragma unroll
  for (int mt = 0; mt < 2; ++mt) {
    int rowBase = mBase + waveM * 32 + mt * 16 + half * 8;
#pragma unroll
    for (int nt = 0; nt < 4; ++nt) {
      int col = nBase + waveN * 64 + nt * 16 + l16;  // 0..1023
      float bv0 = bias[col];
      int h = col >> 6, dh = col & 63;
#pragma unroll
      for (int r = 0; r < 8; ++r) {
        int row = rowBase + r;                        // 0..8191
        int b = row >> 11, s = row & (ATT_S - 1);
        float v = acc[mt][nt][r] + bv0;
        if (mat == 0) {
          v *= 0.125f;  // 1/sqrt(Dh) folded into Q
          q_ws[((size_t)(b * ATT_H + h) * ATT_S + s) * ATT_DH + dh] = f32_to_bf16(v);
        } else if (mat == 1) {
          k_ws[((size_t)(b * ATT_H + h) * ATT_S + s) * ATT_DH + dh] = f32_to_bf16(v);
        } else {
          vt_ws[((size_t)(b * ATT_H + h) * ATT_DH + dh) * ATT_S + s] = f32_to_bf16(v);
        }
      }
    }
  }
}

// ======================= Kernel 2: causal flash attention ==================
// grid = (S/128, B*H). 8 independent waves/block; each wave owns 16 query
// rows, streams 128-wide key tiles with online softmax. No block barriers.
__global__ __launch_bounds__(256) void flash_attn_kernel(
    const unsigned short* __restrict__ q_ws,
    const unsigned short* __restrict__ k_ws,
    const unsigned short* __restrict__ vt_ws,
    unsigned short* __restrict__ attn_ws)   // (B,S,D) bf16
{
  __shared__ unsigned short lP[8][16 * 128];  // per-wave P re-layout buffer

  const int tid  = threadIdx.x;
  const int lane = tid & 31;
  const int wave = tid >> 5;
  const int half = lane >> 4;
  const int l16  = lane & 15;
  const int bh   = blockIdx.y;                 // 0..63
  const int qBase = blockIdx.x * 128 + wave * 16;

  const size_t headOff = (size_t)bh * ATT_S * ATT_DH;
  const unsigned short* Q  = q_ws  + headOff;
  const unsigned short* K  = k_ws  + headOff;
  const unsigned short* Vt = vt_ws + headOff;  // [Dh][S]

  // Q fragments for this wave's 16 rows (kept in registers all kernel)
  bf16x16 aq[2];
  {
    const unsigned short* pq = Q + (size_t)(qBase + l16) * ATT_DH;
    const int kb = half * 16;
    aq[0] = make_frag(*(const uint4*)(pq + kb),      *(const uint4*)(pq + kb + 8));
    aq[1] = make_frag(*(const uint4*)(pq + 32 + kb), *(const uint4*)(pq + 32 + kb + 8));
  }

  floatx8 acc[4];
#pragma unroll
  for (int i = 0; i < 4; ++i) acc[i] = (floatx8){0,0,0,0,0,0,0,0};
  float mrow[8], lrow[8];
#pragma unroll
  for (int r = 0; r < 8; ++r) { mrow[r] = -1e30f; lrow[r] = 0.f; }

  unsigned short* pstage = &lP[wave][0];
  const int jTiles = qBase / 128 + 1;          // causal: up to diag tile

  for (int jt = 0; jt < jTiles; ++jt) {
    const int keyBase = jt * 128;
    floatx8 sf[8];
    // ---- S = Q K^T (scale already in Q) ----
#pragma unroll
    for (int nt = 0; nt < 8; ++nt) {
      const unsigned short* pk =
          K + (size_t)(keyBase + nt * 16 + l16) * ATT_DH + half * 16;
      bf16x16 b0 = make_frag(*(const uint4*)(pk),      *(const uint4*)(pk + 8));
      bf16x16 b1 = make_frag(*(const uint4*)(pk + 32), *(const uint4*)(pk + 40));
      floatx8 c = (floatx8){0,0,0,0,0,0,0,0};
      c = wmma_bf16(aq[0], b0, c);
      c = wmma_bf16(aq[1], b1, c);
      sf[nt] = c;
    }
    // ---- causal mask + running row max ----
    float nm[8];
#pragma unroll
    for (int r = 0; r < 8; ++r) nm[r] = mrow[r];
#pragma unroll
    for (int nt = 0; nt < 8; ++nt) {
      int key = keyBase + nt * 16 + l16;
#pragma unroll
      for (int r = 0; r < 8; ++r) {
        int qrow = qBase + half * 8 + r;
        float v = sf[nt][r];
        if (key > qrow) v = -1e30f;
        sf[nt][r] = v;
        nm[r] = fmaxf(nm[r], v);
      }
    }
#pragma unroll
    for (int off = 1; off < 16; off <<= 1)
#pragma unroll
      for (int r = 0; r < 8; ++r)
        nm[r] = fmaxf(nm[r], __shfl_xor(nm[r], off, 32));

    float alpha[8], rs[8];
#pragma unroll
    for (int r = 0; r < 8; ++r) {
      alpha[r] = __expf(mrow[r] - nm[r]);
      mrow[r]  = nm[r];
      rs[r]    = 0.f;
    }
    // ---- P = exp(S - m), row sums, write P(bf16) to LDS in C layout ----
#pragma unroll
    for (int nt = 0; nt < 8; ++nt) {
      int colBase = nt * 16 + l16;
#pragma unroll
      for (int r = 0; r < 8; ++r) {
        float p = __expf(sf[nt][r] - nm[r]);
        rs[r] += p;
        pstage[(half * 8 + r) * 128 + colBase] = f32_to_bf16(p);
      }
    }
#pragma unroll
    for (int off = 1; off < 16; off <<= 1)
#pragma unroll
      for (int r = 0; r < 8; ++r)
        rs[r] += __shfl_xor(rs[r], off, 32);
#pragma unroll
    for (int r = 0; r < 8; ++r) lrow[r] = lrow[r] * alpha[r] + rs[r];
#pragma unroll
    for (int nt2 = 0; nt2 < 4; ++nt2)
#pragma unroll
      for (int r = 0; r < 8; ++r) acc[nt2][r] = acc[nt2][r] * alpha[r];

    // ---- O += P @ V  (A from LDS re-layout, B from transposed V) ----
    // Same-wave DS ops are in-order: no barrier needed between write & read.
#pragma unroll
    for (int kk = 0; kk < 4; ++kk) {
      const unsigned short* pp = pstage + (size_t)l16 * 128 + kk * 32 + half * 16;
      bf16x16 ap = make_frag(*(const uint4*)pp, *(const uint4*)(pp + 8));
#pragma unroll
      for (int nt2 = 0; nt2 < 4; ++nt2) {
        const unsigned short* pv =
            Vt + (size_t)(nt2 * 16 + l16) * ATT_S + keyBase + kk * 32 + half * 16;
        bf16x16 bv = make_frag(*(const uint4*)pv, *(const uint4*)(pv + 8));
        acc[nt2] = wmma_bf16(ap, bv, acc[nt2]);
      }
    }
  }

  // ---- normalize and store to (B,S,D) bf16 ----
  const int b = bh >> 4, h = bh & 15;
#pragma unroll
  for (int nt2 = 0; nt2 < 4; ++nt2) {
    int d = h * ATT_DH + nt2 * 16 + l16;
#pragma unroll
    for (int r = 0; r < 8; ++r) {
      int s = qBase + half * 8 + r;
      float v = acc[nt2][r] / lrow[r];
      attn_ws[(size_t)(b * ATT_S + s) * ATT_D + d] = f32_to_bf16(v);
    }
  }
}

// ======================= Kernel 3: output projection =======================
// out = attn(bf16) @ Wo(->bf16) + bo, fp32 out. A tile staged with
// GLOBAL_LOAD_ASYNC_TO_LDS_B128 (ASYNCcnt) when the builtin is available.
__global__ __launch_bounds__(256) void out_gemm_kernel(
    const unsigned short* __restrict__ attn,  // (8192,1024) bf16
    const float* __restrict__ Wo, const float* __restrict__ bo,
    float* __restrict__ out)
{
  __shared__ unsigned short lAs[128 * LDSTR];
  __shared__ unsigned short lBs[128 * LDSTR];

  const int tid   = threadIdx.x;
  const int lane  = tid & 31;
  const int wave  = tid >> 5;
  const int waveM = wave & 3;
  const int waveN = wave >> 2;
  const int half  = lane >> 4;
  const int l16   = lane & 15;
  const int mBase = blockIdx.x * 128;
  const int nBase = blockIdx.y * 128;

  floatx8 acc[2][4];
#pragma unroll
  for (int i = 0; i < 2; ++i)
#pragma unroll
    for (int j = 0; j < 4; ++j) acc[i][j] = (floatx8){0,0,0,0,0,0,0,0};

  const int rowB0 = tid >> 5, c4B = (tid & 31) << 2;

  for (int k0 = 0; k0 < ATT_D; k0 += 32) {
    // ---- A tile: 128x32 bf16 = 512 16B chunks, 2 per thread (async) ----
#pragma unroll
    for (int i = 0; i < 2; ++i) {
      int idx = tid + 256 * i;            // 0..511
      int row = idx >> 2;
      int c8  = (idx & 3) << 3;
      async_copy_b128(attn + (size_t)(mBase + row) * ATT_D + k0 + c8,
                      &lAs[row * LDSTR + c8]);
    }
    // ---- B tile: Wo f32 -> bf16, transposed ----
#pragma unroll
    for (int i = 0; i < 4; ++i) {
      int idx = tid + 256 * i;
      int row = idx >> 5;
      int c4  = (idx & 31) << 2;
      float4 f = *(const float4*)(Wo + (size_t)(k0 + row) * ATT_D + nBase + c4);
      lBs[(c4 + 0) * LDSTR + row] = f32_to_bf16(f.x);
      lBs[(c4 + 1) * LDSTR + row] = f32_to_bf16(f.y);
      lBs[(c4 + 2) * LDSTR + row] = f32_to_bf16(f.z);
      lBs[(c4 + 3) * LDSTR + row] = f32_to_bf16(f.w);
    }
    if (k0 + 32 < ATT_D)
      __builtin_prefetch(Wo + (size_t)(k0 + 32 + rowB0) * ATT_D + nBase + c4B, 0, 1);
    wait_async_zero();   // our async LDS writes complete before the barrier
    __syncthreads();

    bf16x16 af[2], bf[4];
    const int kbA = half * 16;
#pragma unroll
    for (int mt = 0; mt < 2; ++mt) {
      const unsigned short* pa = &lAs[(waveM * 32 + mt * 16 + l16) * LDSTR];
      af[mt] = make_frag(*(const uint4*)(pa + kbA),
                         *(const uint4*)(pa + kbA + 8));
    }
    const int kbB = half * 16;
#pragma unroll
    for (int nt = 0; nt < 4; ++nt) {
      const unsigned short* pb = &lBs[(waveN * 64 + nt * 16 + l16) * LDSTR + kbB];
      bf[nt] = make_frag(*(const uint4*)pb, *(const uint4*)(pb + 8));
    }
#pragma unroll
    for (int mt = 0; mt < 2; ++mt)
#pragma unroll
      for (int nt = 0; nt < 4; ++nt)
        acc[mt][nt] = wmma_bf16(af[mt], bf[nt], acc[mt][nt]);
    __syncthreads();
  }

#pragma unroll
  for (int mt = 0; mt < 2; ++mt) {
    int rowBase = mBase + waveM * 32 + mt * 16 + half * 8;
#pragma unroll
    for (int nt = 0; nt < 4; ++nt) {
      int col = nBase + waveN * 64 + nt * 16 + l16;
      float bv0 = bo[col];
#pragma unroll
      for (int r = 0; r < 8; ++r)
        out[(size_t)(rowBase + r) * ATT_D + col] = acc[mt][nt][r] + bv0;
    }
  }
}

// =========================== host-side launcher ============================
extern "C" void kernel_launch(void* const* d_in, const int* in_sizes, int n_in,
                              void* d_out, int out_size, void* d_ws, size_t ws_size,
                              hipStream_t stream) {
  const float* x  = (const float*)d_in[0];
  const float* Wq = (const float*)d_in[1];
  const float* bq = (const float*)d_in[2];
  const float* Wk = (const float*)d_in[3];
  const float* bk = (const float*)d_in[4];
  const float* Wv = (const float*)d_in[5];
  const float* bv = (const float*)d_in[6];
  const float* Wo = (const float*)d_in[7];
  const float* bo = (const float*)d_in[8];
  float* out = (float*)d_out;

  const size_t elems = (size_t)ROWS * ATT_D;              // 8388608
  unsigned short* q_ws    = (unsigned short*)d_ws;        // 16.8 MB each
  unsigned short* k_ws    = q_ws  + elems;
  unsigned short* vt_ws   = k_ws  + elems;
  unsigned short* attn_ws = vt_ws + elems;                // total 67 MB

  dim3 blk(256);
  qkv_gemm_kernel<<<dim3(ROWS / 128, 3 * ATT_D / 128), blk, 0, stream>>>(
      x, Wq, bq, Wk, bk, Wv, bv, q_ws, k_ws, vt_ws);
  flash_attn_kernel<<<dim3(ATT_S / 128, ATT_B * ATT_H), blk, 0, stream>>>(
      q_ws, k_ws, vt_ws, attn_ws);
  out_gemm_kernel<<<dim3(ROWS / 128, ATT_D / 128), blk, 0, stream>>>(
      attn_ws, Wo, bo, out);
}
